// MultiHeadAttention_New_26817775796257
// MI455X (gfx1250) — matrix operand order repaired and checked by reading, not compiled, not run on hardware
//
#include <hip/hip_runtime.h>
#include <hip/hip_bf16.h>

// ---------------------------------------------------------------------------
// MultiHeadAttention forward for MI455X (gfx1250, wave32, WMMA).
// Fixed shapes: B=2, T=S=2048, E=1024, H=16, D=64.
// Pipeline:
//   f32->f16 convert -> Q/K/V projections (WMMA f16, async-LDS weight tiles)
//   -> flash attention (WMMA f16, async-LDS K/V tiles, online softmax)
//   -> output projection (WMMA f16, f32 store).
// CDNA5 paths used: v_wmma_f32_16x16x32_f16, global_load_async_to_lds_b128
// (+ s_wait_asynccnt), ds bounce with s_wait_dscnt, global_prefetch_b8.
// sched_barrier(0) splits fragment-load groups from WMMA groups so loads get
// distinct registers and waits become progressive instead of wait-to-zero.
// ---------------------------------------------------------------------------

typedef __attribute__((ext_vector_type(16))) _Float16 v16h;
typedef __attribute__((ext_vector_type(8)))  _Float16 v8h;
typedef __attribute__((ext_vector_type(8)))  float    v8f;

#define BATCH 2
#define TLEN  2048
#define SLEN  2048
#define EMB   1024
#define NHEAD 16
#define HDIM  64
#define NTOK  (BATCH * TLEN)      // 4096 rows for all projection GEMMs

__device__ __forceinline__ v8f wmma_f16(v16h a, v16h b, v8f c) {
  // D = A(16x32 f16) * B(32x16 f16) + C(16x16 f32)
  return __builtin_amdgcn_wmma_f32_16x16x32_f16(
      /*neg_a=*/false, a, /*neg_b=*/false, b,
      /*c_mod=*/(short)0, c, /*reuse_a=*/false, /*reuse_b=*/false);
}

__device__ __forceinline__ v16h pack16(v8h lo, v8h hi) {
  return __builtin_shufflevector(lo, hi, 0, 1, 2, 3, 4, 5, 6, 7,
                                         8, 9, 10, 11, 12, 13, 14, 15);
}

__device__ __forceinline__ v16h load_frag_lds(const _Float16* p) {
  v8h lo = *(const v8h*)p;
  v8h hi = *(const v8h*)(p + 8);
  return pack16(lo, hi);
}

// One per-lane 16B async global->LDS copy (ASYNCcnt-tracked DMA).
__device__ __forceinline__ void async_b128_to_lds(const _Float16* g,
                                                  const _Float16* l) {
  unsigned lds = (unsigned)(size_t)l;              // low 32b of flat = LDS off
  unsigned long long ga = (unsigned long long)(size_t)g;
  asm volatile("global_load_async_to_lds_b128 %0, %1, off"
               :: "v"(lds), "v"(ga) : "memory");
}

__device__ __forceinline__ void wait_async0() {
  asm volatile("s_wait_asynccnt 0x0" ::: "memory");
}

// ---------------------------------------------------------------------------
// f32 -> f16 elementwise convert (grid-stride)
// ---------------------------------------------------------------------------
__global__ void cvt_f32_f16(const float* __restrict__ src,
                            _Float16* __restrict__ dst, int n) {
  int i = blockIdx.x * blockDim.x + threadIdx.x;
  int stride = gridDim.x * blockDim.x;
  for (; i < n; i += stride) dst[i] = (_Float16)src[i];
}

// ---------------------------------------------------------------------------
// Y[m][n] = (sum_k X16[m][k] * W16[n][k] + bias[n]) * scale
// M = 4096 (tokens), N = 1024, K = 1024. Wave computes a 16x64 tile,
// 8 waves/block -> block tile 128x64. Grid (M/128, N/64) = (32, 16).
// The 64x32 weight slice per K-step is shared by all 8 waves -> staged in
// LDS with double-buffered async loads (1 b128 per thread per step).
// MODE 0: f16 row-major store   (Q, K)
// MODE 1: f16 transposed-per-batch store: Vt[(b*1024+n)*2048 + s]  (V)
// MODE 2: f32 row-major store   (final output projection)
// ---------------------------------------------------------------------------
#define WT_STRIDE 40   // f16 elems per LDS row (80B): 16B-aligned, 2-way min

template <int MODE>
__global__ void __launch_bounds__(256)
gemm_xwt(const _Float16* __restrict__ X, const _Float16* __restrict__ W,
         const float* __restrict__ bias, float scale,
         _Float16* __restrict__ O16, float* __restrict__ O32) {
  __shared__ _Float16 Wt[2][64 * WT_STRIDE];

  const int K = EMB;
  const int tid  = threadIdx.x;
  const int lane = tid & 31;
  const int wave = tid >> 5;
  const int g  = lane >> 4;   // lane half (selects K sub-range)
  const int nl = lane & 15;   // N column within fragment / A row
  const int m0 = blockIdx.x * 128 + wave * 16;
  const int n0 = blockIdx.y * 64;

  // Cooperative async stage of the 64x32 weight slice at k0 into buf.
  // thread -> (row = tid>>2, 16B part = tid&3); 256 threads cover 4KB.
  const int wrow  = tid >> 2;
  const int wpart = tid & 3;
  const _Float16* wsrc_row = W + (size_t)(n0 + wrow) * K + wpart * 8;
  _Float16* wdst_row = &Wt[0][0] + wrow * WT_STRIDE + wpart * 8;

  async_b128_to_lds(wsrc_row /*k0=0*/, wdst_row /*buf0*/);

  v8f acc[4];
  const v8f vz = {0.f, 0.f, 0.f, 0.f, 0.f, 0.f, 0.f, 0.f};
#pragma unroll
  for (int j = 0; j < 4; ++j) acc[j] = vz;

  const _Float16* xrow = X + (size_t)(m0 + nl) * K;

  for (int k0 = 0; k0 < K; k0 += 32) {
    const int cur = (k0 >> 5) & 1;
    wait_async0();       // my slice of tile `cur` has landed
    __syncthreads();     // everyone's slice landed; prev buffer reads done
    if (k0 + 32 < K) {   // issue next tile into the other buffer
      async_b128_to_lds(wsrc_row + (k0 + 32),
                        &Wt[cur ^ 1][0] + wrow * WT_STRIDE + wpart * 8);
    }

    // A fragment: row = m0+nl, K slice kbase = k0 + 16*g, 16 contiguous
    v8h alo = *(const v8h*)(xrow + k0 + 16 * g);
    v8h ahi = *(const v8h*)(xrow + k0 + 16 * g + 8);
    v16h a = pack16(alo, ahi);
    if (k0 + 32 < K) {
      __builtin_prefetch(xrow + k0 + 32, 0, 1);   // global_prefetch_b8
    }

    // Batch all B-fragment loads, then run the WMMA chain. The scheduling
    // barrier keeps the loads (and their dest regs) ahead of all WMMAs.
    v16h bf[4];
#pragma unroll
    for (int j = 0; j < 4; ++j) {
      bf[j] = load_frag_lds(&Wt[cur][0] + (16 * j + nl) * WT_STRIDE + 16 * g);
    }
    __builtin_amdgcn_sched_barrier(0);
#pragma unroll
    for (int j = 0; j < 4; ++j) {
      acc[j] = wmma_f16(a, bf[j], acc[j]);
    }
  }

  // Epilogue: C/D layout -> lane holds (row = m0 + r + 8g, col = n0+16j+nl)
#pragma unroll
  for (int j = 0; j < 4; ++j) {
    const int n = n0 + 16 * j + nl;
    const float bv = bias[n];
#pragma unroll
    for (int r = 0; r < 8; ++r) {
      const int m = m0 + 8 * g + r;
      const float vv = (acc[j][r] + bv) * scale;
      if (MODE == 2) {
        O32[(size_t)m * EMB + n] = vv;
      } else if (MODE == 0) {
        O16[(size_t)m * EMB + n] = (_Float16)vv;
      } else {
        const int b = m >> 11;          // m / SLEN
        const int s = m & (SLEN - 1);   // m % SLEN
        O16[((size_t)(b * EMB + n)) * SLEN + s] = (_Float16)vv;
      }
    }
  }
}

// ---------------------------------------------------------------------------
// Flash attention: one block = (64 query rows, 1 head, 1 batch), 4 waves.
// Each wave owns 16 query rows. S swept in chunks of 64.
// K-tile (64 s-rows x 64 d) and Vt-tile (64 dv-rows x 64 s) are shared by
// all 4 waves -> double-buffered async LDS staging (4 b128 per thread each).
// Q16/K16: [B*len][1024] row-major. Vt16: [B][1024][S] pre-transposed.
// ---------------------------------------------------------------------------
#define KT_STRIDE 72   // f16 elems per LDS row (144B): 16B-aligned
#define PSTRIDE   72

__device__ __forceinline__ void issue_tile64(const _Float16* gbase,
                                             size_t grow_stride,
                                             _Float16* lbase, int tid) {
  // 64 rows x 128B staged as 512 x 16B parts by 128 threads (4 each).
#pragma unroll
  for (int i = 0; i < 4; ++i) {
    const int linear = tid + i * 128;
    const int row  = linear >> 3;
    const int part = linear & 7;
    async_b128_to_lds(gbase + (size_t)row * grow_stride + part * 8,
                      lbase + row * KT_STRIDE + part * 8);
  }
}

__global__ void __launch_bounds__(128)
attn_fa(const _Float16* __restrict__ Q, const _Float16* __restrict__ Km,
        const _Float16* __restrict__ Vt, const int* __restrict__ amask,
        _Float16* __restrict__ O) {
  __shared__ _Float16 Kt[2][64 * KT_STRIDE];
  __shared__ _Float16 Vs[2][64 * KT_STRIDE];
  __shared__ _Float16 Ps[4][16 * PSTRIDE];   // per-wave P bounce buffer

  const int tid  = threadIdx.x;
  const int lane = tid & 31;
  const int wave = tid >> 5;
  const int g  = lane >> 4;
  const int nl = lane & 15;
  const int t0 = blockIdx.x * 64;
  const int h  = blockIdx.y;
  const int b  = blockIdx.z;

  // Tile bases: K rows are s (stride EMB), Vt rows are dv (stride SLEN).
  const _Float16* kbase0 = Km + ((size_t)(b * SLEN)) * EMB + h * HDIM;
  const _Float16* vbase0 = Vt + ((size_t)(b * EMB + h * HDIM)) * SLEN;

  issue_tile64(kbase0 /*s0=0*/, EMB, &Kt[0][0], tid);
  issue_tile64(vbase0 /*s0=0*/, SLEN, &Vs[0][0], tid);

  // Q A-fragments (fixed for the whole S sweep): 2 slices of k (=d) 32 wide.
  const _Float16* qbase =
      Q + ((size_t)(b * TLEN + t0 + wave * 16 + nl)) * EMB + h * HDIM;
  v16h qa[2];
#pragma unroll
  for (int kk = 0; kk < 2; ++kk) {
    v8h lo = *(const v8h*)(qbase + kk * 32 + 16 * g);
    v8h hi = *(const v8h*)(qbase + kk * 32 + 16 * g + 8);
    qa[kk] = pack16(lo, hi);
  }

  // Online-softmax state per accumulator row r (row = r + 8*g, duplicated
  // across the 16 lanes of each half-wave).
  float mrow[8], lrow[8];
  v8f oacc[4];
  const v8f vz = {0.f, 0.f, 0.f, 0.f, 0.f, 0.f, 0.f, 0.f};
#pragma unroll
  for (int r = 0; r < 8; ++r) { mrow[r] = -3.0e38f; lrow[r] = 0.f; }
#pragma unroll
  for (int j = 0; j < 4; ++j) oacc[j] = vz;

  for (int s0 = 0; s0 < SLEN; s0 += 64) {
    const int cur = (s0 >> 6) & 1;
    wait_async0();       // my async slices of tile `cur` landed
    __syncthreads();     // everyone's landed; prev-buffer reads all done
    if (s0 + 64 < SLEN) {
      issue_tile64(kbase0 + (size_t)(s0 + 64) * EMB, EMB, &Kt[cur ^ 1][0], tid);
      issue_tile64(vbase0 + (s0 + 64), SLEN, &Vs[cur ^ 1][0], tid);
    }

    // ---- scores = Q K^T for 16x64 strip (B frags from LDS K tile) ----
    // Batch all 8 fragment loads, then run the WMMA chain.
    v16h kf[8];
#pragma unroll
    for (int kk = 0; kk < 2; ++kk) {
#pragma unroll
      for (int j = 0; j < 4; ++j) {
        // B frag: N = s_local = 16j+nl, k(=d) = kk*32 + 16g + [0..15]
        kf[kk * 4 + j] = load_frag_lds(
            &Kt[cur][0] + (16 * j + nl) * KT_STRIDE + kk * 32 + 16 * g);
      }
    }
    __builtin_amdgcn_sched_barrier(0);
    v8f sc[4];
#pragma unroll
    for (int j = 0; j < 4; ++j) sc[j] = vz;
#pragma unroll
    for (int kk = 0; kk < 2; ++kk) {
#pragma unroll
      for (int j = 0; j < 4; ++j) {
        sc[j] = wmma_f16(qa[kk], kf[kk * 4 + j], sc[j]);
      }
    }

    // ---- clip + additive mask (per column s) ----
    float madd[4];
#pragma unroll
    for (int j = 0; j < 4; ++j) {
      const int mv = amask[b * SLEN + s0 + 16 * j + nl];
      madd[j] = (mv == 0) ? -9.0e15f : (float)mv;
    }
#pragma unroll
    for (int j = 0; j < 4; ++j) {
#pragma unroll
      for (int r = 0; r < 8; ++r) {
        float v = sc[j][r];
        v = fminf(fmaxf(v, -50000.f), 50000.f) + madd[j];
        sc[j][r] = v;
      }
    }

    // ---- online softmax (row reductions across the 16-lane N groups) ----
#pragma unroll
    for (int r = 0; r < 8; ++r) {
      float mx = fmaxf(fmaxf(sc[0][r], sc[1][r]), fmaxf(sc[2][r], sc[3][r]));
#pragma unroll
      for (int d = 1; d < 16; d <<= 1) mx = fmaxf(mx, __shfl_xor(mx, d, 32));
      const float mnew  = fmaxf(mrow[r], mx);
      const float alpha = __expf(mrow[r] - mnew);
      mrow[r] = mnew;
      float rs = 0.f;
#pragma unroll
      for (int j = 0; j < 4; ++j) {
        const float p = __expf(sc[j][r] - mnew);
        sc[j][r] = p;
        rs += p;
      }
#pragma unroll
      for (int d = 1; d < 16; d <<= 1) rs += __shfl_xor(rs, d, 32);
      lrow[r] = lrow[r] * alpha + rs;
#pragma unroll
      for (int j = 0; j < 4; ++j) oacc[j][r] *= alpha;
    }

    // ---- P: C layout -> LDS -> A layout (wave-private buffer) ----
#pragma unroll
    for (int j = 0; j < 4; ++j) {
#pragma unroll
      for (int r = 0; r < 8; ++r) {
        Ps[wave][(r + 8 * g) * PSTRIDE + 16 * j + nl] = (_Float16)sc[j][r];
      }
    }
    asm volatile("s_wait_dscnt 0" ::: "memory");   // lanes read each other's P

    // ---- O += P @ V (B frags from LDS V^T tile) ----
    // Batch P A-frags and all 8 V B-frags, then run the WMMA chain.
    v16h pa[2];
#pragma unroll
    for (int kk = 0; kk < 2; ++kk) {
      pa[kk] = load_frag_lds(&Ps[wave][nl * PSTRIDE + kk * 32 + 16 * g]);
    }
    v16h vf[8];
#pragma unroll
    for (int kk = 0; kk < 2; ++kk) {
#pragma unroll
      for (int j = 0; j < 4; ++j) {
        // B frag: row (= dv) = 16j+nl, k(=s_local) = kk*32 + 16g + [0..15]
        vf[kk * 4 + j] = load_frag_lds(
            &Vs[cur][0] + (16 * j + nl) * KT_STRIDE + kk * 32 + 16 * g);
      }
    }
    __builtin_amdgcn_sched_barrier(0);
#pragma unroll
    for (int kk = 0; kk < 2; ++kk) {
#pragma unroll
      for (int j = 0; j < 4; ++j) {
        oacc[j] = wmma_f16(pa[kk], vf[kk * 4 + j], oacc[j]);
      }
    }
  }

  // ---- normalize and store O (f16, row-major [B*T][1024]) ----
#pragma unroll
  for (int j = 0; j < 4; ++j) {
#pragma unroll
    for (int r = 0; r < 8; ++r) {
      const float vv = oacc[j][r] / lrow[r];
      const size_t row = (size_t)(b * TLEN + t0 + wave * 16 + 8 * g + r);
      O[row * EMB + h * HDIM + 16 * j + nl] = (_Float16)vv;
    }
  }
}

// ---------------------------------------------------------------------------
// Host-side orchestration
// ---------------------------------------------------------------------------
extern "C" void kernel_launch(void* const* d_in, const int* in_sizes, int n_in,
                              void* d_out, int out_size, void* d_ws,
                              size_t ws_size, hipStream_t stream) {
  const float* master = (const float*)d_in[0];
  const float* slave  = (const float*)d_in[1];
  const int*   amask  = (const int*)d_in[2];
  const float* Wq = (const float*)d_in[3];
  const float* bq = (const float*)d_in[4];
  const float* Wk = (const float*)d_in[5];
  const float* bk = (const float*)d_in[6];
  const float* Wv = (const float*)d_in[7];
  const float* bv = (const float*)d_in[8];
  const float* Wo = (const float*)d_in[9];
  const float* bo = (const float*)d_in[10];
  float* out = (float*)d_out;

  // Workspace carving (f16 scratch, ~56 MB total)
  const size_t MT = (size_t)NTOK * EMB;   // 4096*1024
  const size_t WW = (size_t)EMB * EMB;    // 1024*1024
  char* p = (char*)d_ws;
  _Float16* M16  = (_Float16*)p; p += MT * 2;
  _Float16* S16  = (_Float16*)p; p += MT * 2;
  _Float16* Wq16 = (_Float16*)p; p += WW * 2;
  _Float16* Wk16 = (_Float16*)p; p += WW * 2;
  _Float16* Wv16 = (_Float16*)p; p += WW * 2;
  _Float16* Wo16 = (_Float16*)p; p += WW * 2;
  _Float16* Q16  = (_Float16*)p; p += MT * 2;
  _Float16* K16  = (_Float16*)p; p += MT * 2;
  _Float16* Vt16 = (_Float16*)p; p += MT * 2;
  _Float16* O16  = (_Float16*)p; p += MT * 2;

  // 1) f32 -> f16 conversions
  cvt_f32_f16<<<2048, 256, 0, stream>>>(master, M16, (int)MT);
  cvt_f32_f16<<<2048, 256, 0, stream>>>(slave, S16, (int)MT);
  cvt_f32_f16<<<1024, 256, 0, stream>>>(Wq, Wq16, (int)WW);
  cvt_f32_f16<<<1024, 256, 0, stream>>>(Wk, Wk16, (int)WW);
  cvt_f32_f16<<<1024, 256, 0, stream>>>(Wv, Wv16, (int)WW);
  cvt_f32_f16<<<1024, 256, 0, stream>>>(Wo, Wo16, (int)WW);

  // 2) projections
  dim3 ggrid(NTOK / 128, EMB / 64);   // (32, 16)
  const float qscale = 0.125f;        // HEAD_DIM^-0.5 = 1/8
  gemm_xwt<0><<<ggrid, 256, 0, stream>>>(M16, Wq16, bq, qscale, Q16, nullptr);
  gemm_xwt<0><<<ggrid, 256, 0, stream>>>(S16, Wk16, bk, 1.0f, K16, nullptr);
  gemm_xwt<1><<<ggrid, 256, 0, stream>>>(S16, Wv16, bv, 1.0f, Vt16, nullptr);

  // 3) flash attention
  dim3 agrid(TLEN / 64, NHEAD, BATCH);   // (32, 16, 2)
  attn_fa<<<agrid, 128, 0, stream>>>(Q16, K16, Vt16, amask, O16);

  // 4) output projection (f32 store to d_out)
  gemm_xwt<2><<<ggrid, 256, 0, stream>>>(O16, Wo16, bo, 1.0f, nullptr, out);
}
